// MessageLayer_17214228922618
// MI455X (gfx1250) — matrix operand order repaired and checked by reading, not compiled, not run on hardware
//
#include <hip/hip_runtime.h>

typedef __attribute__((ext_vector_type(16))) _Float16 v16h;
typedef __attribute__((ext_vector_type(8)))  _Float16 v8h;
typedef __attribute__((ext_vector_type(8)))  float    v8f;

static constexpr int NN    = 262144;   // nodes
static constexpr int MM    = 1048576;  // edges (2^20)
static constexpr int FF    = 64;       // elem feature len
static constexpr int HGN   = 128;      // gate hidden
static constexpr int HMN   = 128;      // msg hidden
static constexpr int HEADS = 3;
#define NEG_SLOPE 0.01f

// ---------- workspace layout (bytes) ----------
// fg1: gate_w1 fragments  : 3 heads * 32 frags * 1024B = 98304
// fm1: msg_w1  fragments  : 98304
// fm2: msg_w2  fragments  : 3 heads * 16 frags * 1024B = 49152
// g_buf (logits/e/attn)   : 3*M*4 = 12582912
// gmax (mapped uint)      : 3*N*4 = 3145728
// denom                   : 3*N*4 = 3145728     total ~18.2 MB
static constexpr size_t OFF_FM1 = 98304;
static constexpr size_t OFF_FM2 = 196608;
static constexpr size_t OFF_G   = 245760;
static constexpr size_t OFF_MAX = OFF_G   + (size_t)HEADS * MM * 4;
static constexpr size_t OFF_DEN = OFF_MAX + (size_t)HEADS * NN * 4;

// monotone float -> uint mapping for atomic max
__device__ __forceinline__ unsigned fmap(float f) {
  unsigned u = __float_as_uint(f);
  return (int)u >= 0 ? (u | 0x80000000u) : ~u;
}
__device__ __forceinline__ float funmap(unsigned u) {
  unsigned b = (u & 0x80000000u) ? (u & 0x7FFFFFFFu) : ~u;
  return __uint_as_float(b);
}

// Build one f16 A-fragment (16x32 K-slice) for this lane.
// Lane layout: row = lane&15, gsel = lane>>4; halves 0..7 -> K = gsel*8 + i,
// halves 8..15 -> K = 16 + gsel*8 + (i-8)   (all relative to this 32-K slice).
__device__ __forceinline__ v16h make_afrag(const float* __restrict__ p, int gsel) {
  const float4* q0 = (const float4*)(p + gsel * 8);
  const float4* q1 = (const float4*)(p + 16 + gsel * 8);
  float4 f0 = q0[0], f1 = q0[1], f2 = q1[0], f3 = q1[1];
  v16h a;
  a[0]  = (_Float16)f0.x; a[1]  = (_Float16)f0.y; a[2]  = (_Float16)f0.z; a[3]  = (_Float16)f0.w;
  a[4]  = (_Float16)f1.x; a[5]  = (_Float16)f1.y; a[6]  = (_Float16)f1.z; a[7]  = (_Float16)f1.w;
  a[8]  = (_Float16)f2.x; a[9]  = (_Float16)f2.y; a[10] = (_Float16)f2.z; a[11] = (_Float16)f2.w;
  a[12] = (_Float16)f3.x; a[13] = (_Float16)f3.y; a[14] = (_Float16)f3.z; a[15] = (_Float16)f3.w;
  return a;
}

// ---------------- init: residual copy + zero scratch ----------------
__global__ void k_init(const float* __restrict__ x, float* __restrict__ out,
                       unsigned* __restrict__ gmax, float* __restrict__ denom) {
  int i = blockIdx.x * blockDim.x + threadIdx.x;   // NN*FF threads
  out[i] = x[i];
  if (i < HEADS * NN) { gmax[i] = 0u; denom[i] = 0.0f; }
}

// ---------------- weight -> f16 B-fragment swizzle ----------------
// B fragment (32x16): lane = col(n) + 16*gsel ; half i<8 -> B[gsel*8 + i][n], i>=8 -> B[16 + gsel*8 + (i-8)][n]
__global__ void k_conv(const float* __restrict__ gw1, const float* __restrict__ mw1,
                       const float* __restrict__ mw2,
                       _Float16* __restrict__ fg1, _Float16* __restrict__ fm1,
                       _Float16* __restrict__ fm2) {
  int lane = threadIdx.x;
  int b = blockIdx.x;                      // 240 fragments total
  const float* W; _Float16* dst; int NC, idx;
  if (b < 96)       { W = gw1; dst = fg1; NC = 128; idx = b; }
  else if (b < 192) { W = mw1; dst = fm1; NC = 128; idx = b - 96; }
  else              { W = mw2; dst = fm2; NC = 64;  idx = b - 192; }
  int ntiles  = NC >> 4;
  int perhead = 4 * ntiles;
  int head  = idx / perhead, rem = idx % perhead;
  int kstep = rem / ntiles,  nt  = rem % ntiles;
  const float* src = W + (size_t)head * 128 * NC + (size_t)nt * 16;
  int n = lane & 15, gsel = lane >> 4;
  v16h v;
#pragma unroll
  for (int i = 0; i < 16; ++i) {
    int k = kstep * 32 + (i < 8 ? gsel * 8 + i : 16 + gsel * 8 + (i - 8));
    v[i] = (_Float16)src[(size_t)k * NC + n];
  }
  *(v16h*)(dst + (size_t)idx * 512 + lane * 16) = v;
}

// ---------------- gate pass: 32 WMMA + dot per head per 16-edge tile ----------------
__global__ void __launch_bounds__(256) k_gate(
    const float* __restrict__ x,
    const int* __restrict__ self_idx, const int* __restrict__ nbr_idx,
    const _Float16* __restrict__ fg1,
    const float* __restrict__ gb1, const float* __restrict__ gw2,
    const float* __restrict__ gb2, float* __restrict__ g_buf) {
  int lane = threadIdx.x & 31;
  int wid  = threadIdx.x >> 5;
  int tile = blockIdx.x * 8 + wid;
  int n = lane & 15, gsel = lane >> 4;
  int e = tile * 16 + n;
  int self = self_idx[e], nbr = nbr_idx[e];

  v16h afrag[4];
  afrag[0] = make_afrag(x + (size_t)self * FF,      gsel);
  afrag[1] = make_afrag(x + (size_t)self * FF + 32, gsel);
  afrag[2] = make_afrag(x + (size_t)nbr  * FF,      gsel);
  afrag[3] = make_afrag(x + (size_t)nbr  * FF + 32, gsel);

  for (int h = 0; h < HEADS; ++h) {
    const _Float16* bw = fg1 + (size_t)h * 32 * 512 + lane * 16;
    v8f acc[8];
#pragma unroll
    for (int nt = 0; nt < 8; ++nt) {
      float bv = gb1[h * HGN + nt * 16 + n];
      v8f t;
#pragma unroll
      for (int j = 0; j < 8; ++j) t[j] = bv;
      acc[nt] = t;
    }
#pragma unroll
    for (int kk = 0; kk < 4; ++kk) {
#pragma unroll
      for (int nt = 0; nt < 8; ++nt) {
        v16h b = *(const v16h*)(bw + (size_t)(kk * 8 + nt) * 512);
        acc[nt] = __builtin_amdgcn_wmma_f32_16x16x32_f16(
            false, afrag[kk], false, b, (short)0, acc[nt], false, false);
      }
    }
    // LeakyReLU + 128->1 gate head (VALU dot + 16-lane xor reduction)
    float part[8];
#pragma unroll
    for (int j = 0; j < 8; ++j) part[j] = 0.0f;
#pragma unroll
    for (int nt = 0; nt < 8; ++nt) {
      float wv = gw2[h * HGN + nt * 16 + n];
#pragma unroll
      for (int j = 0; j < 8; ++j) {
        float v = acc[nt][j];
        v = v >= 0.0f ? v : NEG_SLOPE * v;
        part[j] += v * wv;
      }
    }
#pragma unroll
    for (int s = 1; s < 16; s <<= 1) {
#pragma unroll
      for (int j = 0; j < 8; ++j) part[j] += __shfl_xor(part[j], s, 32);
    }
    if (n == 0) {
      float b2 = gb2[h];
#pragma unroll
      for (int j = 0; j < 8; ++j)
        g_buf[(size_t)h * MM + tile * 16 + gsel * 8 + j] = part[j] + b2;
    }
  }
}

// ---------------- segment max ----------------
__global__ void k_max(const float* __restrict__ g_buf, const int* __restrict__ self_idx,
                      unsigned* __restrict__ gmax) {
  int t = blockIdx.x * blockDim.x + threadIdx.x;   // HEADS*MM threads
  int h = t >> 20, e = t & (MM - 1);
  atomicMax(&gmax[(size_t)h * NN + self_idx[e]], fmap(g_buf[t]));
}

// ---------------- e = w * exp(g - gmax); segment sum ----------------
__global__ void k_exp(float* __restrict__ g_buf, const int* __restrict__ self_idx,
                      const int* __restrict__ nbr_idx, const float* __restrict__ wts,
                      const unsigned* __restrict__ gmax, float* __restrict__ denom) {
  int t = blockIdx.x * blockDim.x + threadIdx.x;
  int h = t >> 20, e = t & (MM - 1);
  int self = self_idx[e];
  float mx = funmap(gmax[(size_t)h * NN + self]);
  float w  = wts[nbr_idx[e]];                      // WEIGHT_POW == 1.0
  float ev = w * __expf(g_buf[t] - mx);
  g_buf[t] = ev;
  atomicAdd(&denom[(size_t)h * NN + self], ev);
}

// ---------------- attn = e / (denom + 1e-10) * (1/HEADS) ----------------
__global__ void k_attn(float* __restrict__ g_buf, const int* __restrict__ self_idx,
                       const float* __restrict__ denom) {
  int t = blockIdx.x * blockDim.x + threadIdx.x;
  int h = t >> 20, e = t & (MM - 1);
  float d = denom[(size_t)h * NN + self_idx[e]];
  g_buf[t] = g_buf[t] / (d + 1e-10f) * (1.0f / (float)HEADS);
}

// ---------------- message pass: 48 WMMA per head per tile + scatter ----------------
__global__ void __launch_bounds__(256) k_msg(
    const float* __restrict__ x,
    const int* __restrict__ self_idx, const int* __restrict__ nbr_idx,
    const _Float16* __restrict__ fm1, const _Float16* __restrict__ fm2,
    const float* __restrict__ mb1, const float* __restrict__ mb2,
    const float* __restrict__ attn, float* __restrict__ out) {
  __shared__ _Float16 lds_h[8][16 * 128];          // 32 KB: per-wave h-tile staging
  int lane = threadIdx.x & 31;
  int wid  = threadIdx.x >> 5;
  int tile = blockIdx.x * 8 + wid;
  int n = lane & 15, gsel = lane >> 4;
  int e = tile * 16 + n;
  int self = self_idx[e], nbr = nbr_idx[e];

  v16h afrag[4];
  afrag[0] = make_afrag(x + (size_t)self * FF,      gsel);
  afrag[1] = make_afrag(x + (size_t)self * FF + 32, gsel);
  afrag[2] = make_afrag(x + (size_t)nbr  * FF,      gsel);
  afrag[3] = make_afrag(x + (size_t)nbr  * FF + 32, gsel);

  int selfm[8];
#pragma unroll
  for (int j = 0; j < 8; ++j) selfm[j] = self_idx[tile * 16 + gsel * 8 + j];

  _Float16* myl = lds_h[wid];

  for (int h = 0; h < HEADS; ++h) {
    // ---- layer 1: 16x128 @ 128x128 ----
    const _Float16* bw = fm1 + (size_t)h * 32 * 512 + lane * 16;
    v8f acc[8];
#pragma unroll
    for (int nt = 0; nt < 8; ++nt) {
      float bv = mb1[h * HMN + nt * 16 + n];
      v8f t;
#pragma unroll
      for (int j = 0; j < 8; ++j) t[j] = bv;
      acc[nt] = t;
    }
#pragma unroll
    for (int kk = 0; kk < 4; ++kk) {
#pragma unroll
      for (int nt = 0; nt < 8; ++nt) {
        v16h b = *(const v16h*)(bw + (size_t)(kk * 8 + nt) * 512);
        acc[nt] = __builtin_amdgcn_wmma_f32_16x16x32_f16(
            false, afrag[kk], false, b, (short)0, acc[nt], false, false);
      }
    }
    // ---- LeakyReLU, C-layout -> row-major f16 in LDS ----
    __syncthreads();
#pragma unroll
    for (int nt = 0; nt < 8; ++nt) {
#pragma unroll
      for (int j = 0; j < 8; ++j) {
        float v = acc[nt][j];
        v = v >= 0.0f ? v : NEG_SLOPE * v;
        myl[(gsel * 8 + j) * 128 + nt * 16 + n] = (_Float16)v;
      }
    }
    __syncthreads();
    // ---- layer 2: 16x128 @ 128x64 (A-fragments re-read from LDS) ----
    v8f acc2[4];
#pragma unroll
    for (int nt = 0; nt < 4; ++nt) {
      float bv = mb2[h * FF + nt * 16 + n];
      v8f t;
#pragma unroll
      for (int j = 0; j < 8; ++j) t[j] = bv;
      acc2[nt] = t;
    }
    const _Float16* bw2 = fm2 + (size_t)h * 16 * 512 + lane * 16;
#pragma unroll
    for (int kk = 0; kk < 4; ++kk) {
      v8h lo = *(const v8h*)(myl + n * 128 + kk * 32 + gsel * 8);
      v8h hi = *(const v8h*)(myl + n * 128 + kk * 32 + 16 + gsel * 8);
      v16h a2;
#pragma unroll
      for (int i = 0; i < 8; ++i) { a2[i] = lo[i]; a2[8 + i] = hi[i]; }
#pragma unroll
      for (int nt = 0; nt < 4; ++nt) {
        v16h b = *(const v16h*)(bw2 + (size_t)(kk * 4 + nt) * 512);
        acc2[nt] = __builtin_amdgcn_wmma_f32_16x16x32_f16(
            false, a2, false, b, (short)0, acc2[nt], false, false);
      }
    }
    // ---- attn scale + scatter-add ----
    float attnv[8];
#pragma unroll
    for (int j = 0; j < 8; ++j)
      attnv[j] = attn[(size_t)h * MM + tile * 16 + gsel * 8 + j];
#pragma unroll
    for (int nt = 0; nt < 4; ++nt) {
#pragma unroll
      for (int j = 0; j < 8; ++j)
        atomicAdd(&out[(size_t)selfm[j] * FF + nt * 16 + n], acc2[nt][j] * attnv[j]);
    }
  }
}

extern "C" void kernel_launch(void* const* d_in, const int* in_sizes, int n_in,
                              void* d_out, int out_size, void* d_ws, size_t ws_size,
                              hipStream_t stream) {
  const float* elem_weights = (const float*)d_in[0];
  const float* x            = (const float*)d_in[1];
  const int*   self_idx     = (const int*)d_in[2];
  const int*   nbr_idx      = (const int*)d_in[3];
  const float* gw1          = (const float*)d_in[4];
  const float* gb1          = (const float*)d_in[5];
  const float* gw2          = (const float*)d_in[6];
  const float* gb2          = (const float*)d_in[7];
  const float* mw1          = (const float*)d_in[8];
  const float* mb1          = (const float*)d_in[9];
  const float* mw2          = (const float*)d_in[10];
  const float* mb2          = (const float*)d_in[11];
  float* out = (float*)d_out;

  char* ws = (char*)d_ws;
  _Float16* fg1   = (_Float16*)(ws);
  _Float16* fm1   = (_Float16*)(ws + OFF_FM1);
  _Float16* fm2   = (_Float16*)(ws + OFF_FM2);
  float*    g_buf = (float*)(ws + OFF_G);
  unsigned* gmax  = (unsigned*)(ws + OFF_MAX);
  float*    denom = (float*)(ws + OFF_DEN);

  k_init<<<(NN * FF) / 256, 256, 0, stream>>>(x, out, gmax, denom);
  k_conv<<<240, 32, 0, stream>>>(gw1, mw1, mw2, fg1, fm1, fm2);
  k_gate<<<MM / 16 / 8, 256, 0, stream>>>(x, self_idx, nbr_idx, fg1, gb1, gw2, gb2, g_buf);
  k_max<<<(HEADS * MM) / 256, 256, 0, stream>>>(g_buf, self_idx, gmax);
  k_exp<<<(HEADS * MM) / 256, 256, 0, stream>>>(g_buf, self_idx, nbr_idx, elem_weights, gmax, denom);
  k_attn<<<(HEADS * MM) / 256, 256, 0, stream>>>(g_buf, self_idx, denom);
  k_msg<<<MM / 16 / 8, 256, 0, stream>>>(x, self_idx, nbr_idx, fm1, fm2, mb1, mb2, g_buf, out);
}
